// SiameseSNN_42142219108391
// MI455X (gfx1250) — matrix operand order repaired and checked by reading, not compiled, not run on hardware
//
#include <hip/hip_runtime.h>

// ---------------- types ----------------
typedef __attribute__((ext_vector_type(16))) _Float16 v16h;
typedef __attribute__((ext_vector_type(8)))  _Float16 v8h;
typedef __attribute__((ext_vector_type(8)))  float    v8f;
typedef __attribute__((ext_vector_type(4)))  unsigned u32x4;
typedef __attribute__((ext_vector_type(8)))  int      i32x8;
typedef __attribute__((ext_vector_type(4)))  int      i32x4;

// ---------------- model dims ----------------
#define BB     32      // batch
#define C1C    16      // conv1 out channels
#define L1L    1981    // conv1 out length ((8000-80)/4+1)
#define LPP    495     // after maxpool4 (VALID)
#define C2C    32      // conv2 out channels
#define L2L    493     // conv2 out length (495-3+1)
#define KFLAT  15776   // 32*493
#define NHID   250
#define NPADD  256     // padded hidden for 16x16 tiling
#define TSTEPS 50

#define BETA   0.9f
#define THR    1.0f
#define BNEPS  1e-5f

// ============ fc1w -> f16, padded [256][15776], rows>=250 zero ============
__global__ void k_cvt_fc1w(const float* __restrict__ w, _Float16* __restrict__ wh) {
  int idx = blockIdx.x * blockDim.x + threadIdx.x;
  int total = NPADD * KFLAT;
  if (idx >= total) return;
  int n = idx / KFLAT;
  wh[idx] = (n < NHID) ? (_Float16)w[idx] : (_Float16)0.0f;
}

// ============ zero state, first-spike = -1 ============
__global__ void k_init(float* __restrict__ mem1, float* __restrict__ mem2,
                       float* __restrict__ mem3, float* __restrict__ fr,
                       float* __restrict__ ft) {
  int idx = blockIdx.x * blockDim.x + threadIdx.x;
  if (idx < BB * C1C * L1L) mem1[idx] = 0.0f;
  if (idx < BB * C2C * L2L) mem2[idx] = 0.0f;
  if (idx < BB * NPADD) { mem3[idx] = 0.0f; fr[idx] = 0.0f; ft[idx] = -1.0f; }
}

// ============ conv1 (stride 4, K=80) + BN1 (eval) -> cur1 ============
__global__ void k_conv1_bn(const float* __restrict__ x, const float* __restrict__ w,
                           const float* __restrict__ cb, const float* __restrict__ g,
                           const float* __restrict__ bt, const float* __restrict__ mn,
                           const float* __restrict__ vr, float* __restrict__ cur1) {
  __shared__ float sw[C1C * 80];
  for (int i = threadIdx.x; i < C1C * 80; i += blockDim.x) sw[i] = w[i];
  __syncthreads();
  int idx = blockIdx.x * blockDim.x + threadIdx.x;
  int total = BB * C1C * L1L;
  if (idx >= total) return;
  int l = idx % L1L;
  int c = (idx / L1L) % C1C;
  int b = idx / (L1L * C1C);
  const float* xr = x + b * 8000 + l * 4;
  const float* wr = sw + c * 80;
  float acc = 0.0f;
#pragma unroll 8
  for (int k = 0; k < 80; ++k) acc += xr[k] * wr[k];
  acc += cb[c];
  float inv = g[c] * rsqrtf(vr[c] + BNEPS);
  cur1[idx] = (acc - mn[c]) * inv + bt[c];
}

// ============ LIF layer-1 (subtract reset) + maxpool4 over spikes ============
__global__ void k_lif1_pool(const float* __restrict__ cur1, float* __restrict__ mem1,
                            float* __restrict__ pooled) {
  int idx = blockIdx.x * blockDim.x + threadIdx.x;
  int total = BB * C1C * LPP;
  if (idx >= total) return;
  int gg = idx % LPP;
  int c = (idx / LPP) % C1C;
  int b = idx / (LPP * C1C);
  int base = (b * C1C + c) * L1L + gg * 4;
  float spkmax = 0.0f;
#pragma unroll
  for (int k = 0; k < 4; ++k) {
    float mem = mem1[base + k];
    float reset = (mem > THR) ? THR : 0.0f;       // reset from previous mem (detached)
    mem = BETA * mem + cur1[base + k] - reset;
    mem1[base + k] = mem;
    spkmax = fmaxf(spkmax, (mem > THR) ? 1.0f : 0.0f);
  }
  pooled[idx] = spkmax;                            // max of 0/1 spikes == OR
}

// ============ conv2 (K=3) + BN2 + LIF layer-2 ============
// One block per batch row. pooled[b] (16x495 f32 = 31.7KB) is staged into LDS by the
// Tensor Data Mover (TENSOR_LOAD_TO_LDS + s_wait_tensorcnt), then 512 threads
// (32 channels x 16 lanes) compute with the 48 per-channel weights held in VGPRs.
__global__ void __launch_bounds__(512) k_conv2_lif2(
    const float* __restrict__ pooled, const float* __restrict__ w2,
    const float* __restrict__ cb, const float* __restrict__ g,
    const float* __restrict__ bt, const float* __restrict__ mn,
    const float* __restrict__ vr, float* __restrict__ mem2,
    _Float16* __restrict__ spk2flat) {
  __shared__ float spool[C1C * LPP];               // 31680 bytes
  const int b = blockIdx.x;
  const float* src = pooled + (size_t)b * C1C * LPP;

#if __has_builtin(__builtin_amdgcn_tensor_load_to_lds)
  if (threadIdx.x == 0) {                          // TDM ignores EXEC; one wave issues it
    const unsigned long long ga = (unsigned long long)(uintptr_t)src;
    const unsigned lds_off = (unsigned)(uintptr_t)(&spool[0]);  // flat low 32b = LDS offset
    const unsigned NEL = C1C * LPP;                // 7920 contiguous f32
    u32x4 g0;
    g0[0] = 1u;                                    // count=1, user-mode descriptor
    g0[1] = lds_off;                               // lds_addr (bytes)
    g0[2] = (unsigned)(ga & 0xFFFFFFFFu);          // global_addr[31:0]
    g0[3] = (unsigned)((ga >> 32) & 0x01FFFFFFu) | (2u << 30);  // addr[56:32] | type=2
    i32x8 g1;
    g1[0] = (int)(2u << 16);                       // workgroup_mask=0, data_size=2 (4B)
    g1[1] = (int)((NEL & 0xFFFFu) << 16);          // tensor_dim0[15:0]
    g1[2] = (int)((NEL >> 16) | (1u << 16));       // tensor_dim0[31:16] | tensor_dim1[15:0]=1
    g1[3] = (int)(NEL << 16);                      // tensor_dim1[31:16]=0 | tile_dim0=NEL
    g1[4] = 1;                                     // tile_dim1=1, tile_dim2=0
    g1[5] = (int)NEL;                              // tensor_dim0_stride[31:0]
    g1[6] = 0;
    g1[7] = 0;
    i32x4 gz = {0, 0, 0, 0};
#if defined(__clang_major__) && (__clang_major__ >= 23)
    i32x8 gz8 = {0, 0, 0, 0, 0, 0, 0, 0};
    __builtin_amdgcn_tensor_load_to_lds(g0, g1, gz, gz, gz8, 0);
#else
    __builtin_amdgcn_tensor_load_to_lds(g0, g1, gz, gz, 0);
#endif
    __builtin_amdgcn_s_wait_tensorcnt(0);
  }
#else
  for (int i = threadIdx.x; i < C1C * LPP; i += (int)blockDim.x) spool[i] = src[i];
#endif
  __syncthreads();

  const int o  = threadIdx.x >> 4;                 // 0..31 output channel
  const int lt = threadIdx.x & 15;                 // 0..15 lane within channel
  float wreg[C1C * 3];
#pragma unroll
  for (int i = 0; i < C1C * 3; ++i) wreg[i] = w2[o * C1C * 3 + i];
  const float inv   = g[o] * rsqrtf(vr[o] + BNEPS);
  const float shift = bt[o] - mn[o] * inv;
  const float biasc = cb[o];

  for (int l = lt; l < L2L; l += 16) {
    float acc = biasc;
#pragma unroll
    for (int i = 0; i < C1C; ++i) {
      const float* p = &spool[i * LPP + l];
      acc += p[0] * wreg[i * 3 + 0] + p[1] * wreg[i * 3 + 1] + p[2] * wreg[i * 3 + 2];
    }
    const float cur = acc * inv + shift;
    const int idx = (b * C2C + o) * L2L + l;
    float mem = mem2[idx];
    const float reset = (mem > THR) ? THR : 0.0f;
    mem = BETA * mem + cur - reset;
    mem2[idx] = mem;
    spk2flat[b * KFLAT + o * L2L + l] = (_Float16)((mem > THR) ? 1.0f : 0.0f);
  }
}

// ============ fc1 via v_wmma_f32_16x16x32_f16 + fused LIF3 / stats ============
// grid = 16 (N tiles of 16, NHID padded to 256), block = 512 (16 waves).
// wave w: Mtile = w&1, contiguous K-range kpart = w>>1 (8-way split).
// Depth-2 software pipeline: loads of chunk k+1 are in flight during WMMA of chunk k.
__global__ void __launch_bounds__(512) k_fc1_wmma_lif3(
    const _Float16* __restrict__ A,   // spikes  [32][15776] f16 (exact 0/1)
    const _Float16* __restrict__ Bw,  // weights [256][15776] f16 (rows>=250 zero)
    const float* __restrict__ fc1b,   // [250]
    float* __restrict__ mem3, float* __restrict__ fr, float* __restrict__ ft,
    int t) {
  __shared__ float cred[16][32][8];   // 16 KB partials
  const int tid   = threadIdx.x;
  const int w     = tid >> 5;    // wave 0..15
  const int ln    = tid & 31;
  const int l15   = ln & 15;
  const int khalf = ln >> 4;
  const int mt    = w & 1;       // M tile
  const int kpart = w >> 1;      // contiguous K split 0..7
  const int Mbase = mt * 16;
  const int Nbase = blockIdx.x * 16;
  const int k0 = kpart * 62;
  const int k1 = (kpart == 7) ? 493 : (k0 + 62);

  // A layout (16-bit A 16x32): lanes 0-15 hold K {0..7,16..23}; lanes 16-31 {8..15,24..31}
  const _Float16* ap = A + (Mbase + l15) * KFLAT + khalf * 8 + k0 * 32;
  // B layout (16-bit B 32x16): lane = column; lanes 0-15 K {0..7,16..23}; lanes 16-31 {8..15,24..31}
  const _Float16* bp = Bw + (Nbase + l15) * KFLAT + khalf * 8 + k0 * 32;

  v8f c = {};
  v8h a0 = *(const v8h*)ap;
  v8h a1 = *(const v8h*)(ap + 16);
  v8h b0 = *(const v8h*)bp;
  v8h b1 = *(const v8h*)(bp + 16);
#pragma unroll 2
  for (int kc = k0 + 1; kc < k1; ++kc) {
    ap += 32; bp += 32;
    __builtin_prefetch(bp + 64, 0, 1);
    __builtin_prefetch(ap + 64, 0, 1);
    v8h na0 = *(const v8h*)ap;
    v8h na1 = *(const v8h*)(ap + 16);
    v8h nb0 = *(const v8h*)bp;
    v8h nb1 = *(const v8h*)(bp + 16);
    v16h av = __builtin_shufflevector(a0, a1, 0,1,2,3,4,5,6,7,8,9,10,11,12,13,14,15);
    v16h bv = __builtin_shufflevector(b0, b1, 0,1,2,3,4,5,6,7,8,9,10,11,12,13,14,15);
    c = __builtin_amdgcn_wmma_f32_16x16x32_f16(false, av, false, bv, (short)0, c,
                                               false, false);
    a0 = na0; a1 = na1; b0 = nb0; b1 = nb1;
  }
  {
    v16h av = __builtin_shufflevector(a0, a1, 0,1,2,3,4,5,6,7,8,9,10,11,12,13,14,15);
    v16h bv = __builtin_shufflevector(b0, b1, 0,1,2,3,4,5,6,7,8,9,10,11,12,13,14,15);
    c = __builtin_amdgcn_wmma_f32_16x16x32_f16(false, av, false, bv, (short)0, c,
                                               false, false);
  }

#pragma unroll
  for (int vi = 0; vi < 8; ++vi) cred[w][ln][vi] = c[vi];
  __syncthreads();

  if (tid < 64) {                         // one lane-set per M tile does the epilogue
    const int emt  = tid >> 5;
    const int eln  = tid & 31;
    const int el15 = eln & 15;
    const int ekh  = eln >> 4;
    const int n = Nbase + el15;
    if (n < NHID) {
      const float bias = fc1b[n];
#pragma unroll
      for (int vi = 0; vi < 8; ++vi) {
        // D layout: lane gives N; VGPR vi gives M=vi (lanes 0-15) / vi+8 (lanes 16-31)
        float cur = bias;
#pragma unroll
        for (int s = 0; s < 8; ++s) cur += cred[emt + 2 * s][eln][vi];
        const int m = emt * 16 + vi + 8 * ekh;
        const int off = m * NPADD + n;
        float mem = mem3[off];
        const float reset = (mem > THR) ? THR : 0.0f;
        mem = BETA * mem + cur - reset;
        mem3[off] = mem;
        if (mem > THR) {                  // spike
          fr[off] += 1.0f;                // firing rate accumulation
          if (ft[off] < 0.0f) ft[off] = (float)t;  // argmax over 0/1 train = first spike
        }
      }
    }
  }
}

// ============ head: comb[32,500] @ fcsw.T + b, L2-normalize rows ============
__global__ void k_head(const float* __restrict__ fr, const float* __restrict__ ft,
                       const float* __restrict__ fcsw, const float* __restrict__ fcsb,
                       float* __restrict__ out) {
  const int b = blockIdx.x;   // 0..31
  const int n = threadIdx.x;  // 0..127
  __shared__ float comb[2 * NHID];
  __shared__ float red[128];
  for (int j = threadIdx.x; j < 2 * NHID; j += blockDim.x) {
    if (j < NHID) comb[j] = fr[b * NPADD + j];
    else {
      float f = ft[b * NPADD + (j - NHID)];
      comb[j] = ((f < 0.0f) ? 0.0f : f) * (1.0f / (float)TSTEPS);
    }
  }
  __syncthreads();
  float acc = fcsb[n];
  const float* wr = fcsw + n * 2 * NHID;
  for (int j = 0; j < 2 * NHID; ++j) acc += comb[j] * wr[j];
  red[n] = acc * acc;
  __syncthreads();
  for (int s = 64; s > 0; s >>= 1) {
    if (n < s) red[n] += red[n + s];
    __syncthreads();
  }
  float norm = fmaxf(sqrtf(red[0]), 1e-12f);
  out[b * 128 + n] = acc / norm;
}

// ============ launch ============
extern "C" void kernel_launch(void* const* d_in, const int* in_sizes, int n_in,
                              void* d_out, int out_size, void* d_ws, size_t ws_size,
                              hipStream_t stream) {
  (void)in_sizes; (void)n_in; (void)out_size; (void)ws_size;
  const float* xin[2] = {(const float*)d_in[0], (const float*)d_in[1]};
  const float* c1w  = (const float*)d_in[2];
  const float* c1b  = (const float*)d_in[3];
  const float* bn1g = (const float*)d_in[4];
  const float* bn1b = (const float*)d_in[5];
  const float* bn1m = (const float*)d_in[6];
  const float* bn1v = (const float*)d_in[7];
  const float* c2w  = (const float*)d_in[8];
  const float* c2b  = (const float*)d_in[9];
  const float* bn2g = (const float*)d_in[10];
  const float* bn2b = (const float*)d_in[11];
  const float* bn2m = (const float*)d_in[12];
  const float* bn2v = (const float*)d_in[13];
  const float* fc1w = (const float*)d_in[14];
  const float* fc1b = (const float*)d_in[15];
  const float* fcsw = (const float*)d_in[16];
  const float* fcsb = (const float*)d_in[17];

  // ---- workspace carve (256B aligned), ~20.3 MB total, reused for both embeddings ----
  char* p = (char*)d_ws;
  auto carve = [&p](size_t bytes) -> char* {
    char* r = p;
    p += (bytes + 255) & ~(size_t)255;
    return r;
  };
  _Float16* fc1wh = (_Float16*)carve((size_t)NPADD * KFLAT * sizeof(_Float16));
  float* cur1     = (float*)carve((size_t)BB * C1C * L1L * sizeof(float));
  float* mem1     = (float*)carve((size_t)BB * C1C * L1L * sizeof(float));
  float* pooled   = (float*)carve((size_t)BB * C1C * LPP * sizeof(float));
  float* mem2     = (float*)carve((size_t)BB * C2C * L2L * sizeof(float));
  _Float16* spk2  = (_Float16*)carve((size_t)BB * KFLAT * sizeof(_Float16));
  float* mem3     = (float*)carve((size_t)BB * NPADD * sizeof(float));
  float* fr       = (float*)carve((size_t)BB * NPADD * sizeof(float));
  float* ft       = (float*)carve((size_t)BB * NPADD * sizeof(float));

  const int n_cvt  = NPADD * KFLAT;
  const int n_cur1 = BB * C1C * L1L;
  const int n_pool = BB * C1C * LPP;

  k_cvt_fc1w<<<(n_cvt + 255) / 256, 256, 0, stream>>>(fc1w, fc1wh);

  for (int e = 0; e < 2; ++e) {
    k_init<<<(n_cur1 + 255) / 256, 256, 0, stream>>>(mem1, mem2, mem3, fr, ft);
    k_conv1_bn<<<(n_cur1 + 255) / 256, 256, 0, stream>>>(xin[e], c1w, c1b, bn1g, bn1b,
                                                         bn1m, bn1v, cur1);
    for (int t = 0; t < TSTEPS; ++t) {
      k_lif1_pool<<<(n_pool + 255) / 256, 256, 0, stream>>>(cur1, mem1, pooled);
      k_conv2_lif2<<<BB, 512, 0, stream>>>(pooled, c2w, c2b, bn2g, bn2b,
                                           bn2m, bn2v, mem2, spk2);
      k_fc1_wmma_lif3<<<16, 512, 0, stream>>>(spk2, fc1wh, fc1b, mem3, fr, ft, t);
    }
    k_head<<<32, 128, 0, stream>>>(fr, ft, fcsw, fcsb, (float*)d_out + (size_t)e * BB * 128);
  }
}
